// HierarchicalQueryAwareGATRetriever_49727131353450
// MI455X (gfx1250) — matrix operand order repaired and checked by reading, not compiled, not run on hardware
//
#include <hip/hip_runtime.h>
#include <hip/hip_bf16.h>

#define TOTAL 4096
#define EMB   384
#define HID   256
#define HEADS 4
#define HD    64
#define MDOCS 32
#define NSENT (TOTAL - 1 - MDOCS)   /* 4063 */
#define FEATK  514
#define FEATKP 544
#define MPAD   4064
#define AWORDS (TOTAL / 32)         /* 128 bitmask words per adjacency row */

typedef __attribute__((ext_vector_type(16))) __bf16 v16bf;
typedef __attribute__((ext_vector_type(8)))  float  v8f;

union FragBF { unsigned int u[8]; v16bf v; };

__device__ __forceinline__ unsigned short f2bf(float f) {
  unsigned int u = __float_as_uint(f);
  u += 0x7FFFu + ((u >> 16) & 1u);          // round-to-nearest-even
  return (unsigned short)(u >> 16);
}
__device__ __forceinline__ float bf2f(unsigned short s) {
  return __uint_as_float(((unsigned int)s) << 16);
}
__device__ __forceinline__ unsigned int pack2bf(float lo, float hi) {
  return (unsigned int)f2bf(lo) | ((unsigned int)f2bf(hi) << 16);
}
__device__ __forceinline__ float lrelu(float x) { return x > 0.f ? x : 0.2f * x; }

// ---------------- converts ----------------
__global__ void k_f32_to_bf16(const float* __restrict__ in, unsigned short* __restrict__ out, int n) {
  int i = blockIdx.x * 256 + threadIdx.x;
  if (i < n) out[i] = f2bf(in[i]);
}

__global__ void k_pad_fusionw(const float* __restrict__ w, unsigned short* __restrict__ out) {
  int i = blockIdx.x * 256 + threadIdx.x;
  if (i >= HID * FEATKP) return;
  int r = i / FEATKP, k = i % FEATKP;
  out[i] = (k < FEATK) ? f2bf(w[r * FEATK + k]) : (unsigned short)0;
}

// pack adjacency into a bitmask: bit j of bits[row*AWORDS + j/32] = (adj[row][j] != 0)
__global__ void k_packadj(const int* __restrict__ adj, unsigned int* __restrict__ bits) {
  int i = blockIdx.x * 256 + threadIdx.x;          // one thread per adj element
  unsigned long long b = __ballot(adj[i] != 0);    // wave32: low 32 bits valid
  if ((threadIdx.x & 31) == 0) bits[i >> 5] = (unsigned int)b;
}

// ---------------- generic WMMA GEMM: C = act(A(MxK) * Bt(NxK)^T + bias) ----------------
// block = 128 threads (4 waves); wave w -> 16x16 tile at (blockIdx.x*16, blockIdx.y*64 + w*16)
__global__ __launch_bounds__(128) void wmma_gemm_bt(
    const unsigned short* __restrict__ A, const unsigned short* __restrict__ Bt,
    const float* __restrict__ bias, float* __restrict__ C, unsigned short* __restrict__ Ct,
    int M, int N, int K, int relu) {
  const int lane = threadIdx.x & 31;
  const int wave = threadIdx.x >> 5;
  const int tm = blockIdx.x * 16;
  const int tn = blockIdx.y * 64 + wave * 16;
  const int m = lane & 15, hi = lane >> 4, n = lane & 15;
  v8f acc = {0.f, 0.f, 0.f, 0.f, 0.f, 0.f, 0.f, 0.f};
  const unsigned short* Arow = A + (size_t)(tm + m) * K;
  const unsigned short* Brow = Bt + (size_t)(tn + n) * K;
  for (int kt = 0; kt < K; kt += 32) {
    FragBF a, b;
#pragma unroll
    for (int v = 0; v < 8; ++v) {
      int ka = kt + 2 * v + ((v >= 4) ? 8 : 0) + 8 * hi;  // A 16x32 bf16 layout
      int kb = kt + 2 * v + ((v >= 4) ? 8 : 0) + 8 * hi;  // B 32x16 bf16 layout (col per lane)
      a.u[v] = *(const unsigned int*)(Arow + ka);
      b.u[v] = *(const unsigned int*)(Brow + kb);
    }
    acc = __builtin_amdgcn_wmma_f32_16x16x32_bf16(false, a.v, false, b.v, (short)0, acc, false, false);
  }
  float bv = bias ? bias[tn + n] : 0.f;
#pragma unroll
  for (int v = 0; v < 8; ++v) {
    int row = tm + v + 8 * hi;                            // C/D: M = vgpr + 8*hi, N = lane&15
    float val = acc[v] + bv;
    if (relu) val = fmaxf(val, 0.f);
    if (C)  C[(size_t)row * N + tn + n] = val;
    if (Ct) Ct[(size_t)(tn + n) * M + row] = f2bf(val);   // transposed bf16 store
  }
}

// ---------------- query norm ----------------
__global__ void k_qnorm(const float* __restrict__ h, float* __restrict__ qn) {
  __shared__ float red[256];
  float v = (threadIdx.x < HID) ? h[threadIdx.x] : 0.f;
  red[threadIdx.x] = v * v;
  __syncthreads();
  for (int o = 128; o; o >>= 1) { if (threadIdx.x < o) red[threadIdx.x] += red[threadIdx.x + o]; __syncthreads(); }
  if (threadIdx.x == 0) *qn = sqrtf(red[0]);
}

__global__ __launch_bounds__(256) void k_blend(float* __restrict__ h, const float* __restrict__ qn) {
  int wave = threadIdx.x >> 5, lane = threadIdx.x & 31;
  int row = 1 + MDOCS + blockIdx.x * 8 + wave;
  if (row >= TOTAL) return;
  float* s = h + (size_t)row * HID;
  const float* q = h;
  float dot = 0.f, sn2 = 0.f;
  for (int c = lane; c < HID; c += 32) { float sv = s[c]; dot += q[c] * sv; sn2 += sv * sv; }
  for (int o = 16; o; o >>= 1) { dot += __shfl_xor(dot, o, 32); sn2 += __shfl_xor(sn2, o, 32); }
  float cosv = dot / ((*qn + 1e-8f) * (sqrtf(sn2) + 1e-8f));
  for (int c = lane; c < HID; c += 32) s[c] += 0.8f * q[c] * cosv;
}

// ---------------- layernorm -> bf16 ----------------
__global__ __launch_bounds__(256) void k_layernorm(const float* __restrict__ h, const float* __restrict__ sc,
                                                   const float* __restrict__ bs, unsigned short* __restrict__ xbf) {
  int wave = threadIdx.x >> 5, lane = threadIdx.x & 31;
  int row = blockIdx.x * 8 + wave;
  const float* x = h + (size_t)row * HID;
  float s = 0.f, s2 = 0.f;
  for (int c = lane; c < HID; c += 32) { float v = x[c]; s += v; s2 += v * v; }
  for (int o = 16; o; o >>= 1) { s += __shfl_xor(s, o, 32); s2 += __shfl_xor(s2, o, 32); }
  float mu = s * (1.f / HID);
  float inv = rsqrtf(s2 * (1.f / HID) - mu * mu + 1e-5f);
  for (int c = lane; c < HID; c += 32)
    xbf[(size_t)row * HID + c] = f2bf((x[c] - mu) * inv * sc[c] + bs[c]);
}

// ---------------- src/dst projections from WhT ----------------
__global__ void k_srcdst(const unsigned short* __restrict__ whT, const float* __restrict__ gat_a,
                         float* __restrict__ src, float* __restrict__ dst) {
  int i = blockIdx.x * 256 + threadIdx.x;
  if (i >= HEADS * TOTAL) return;
  int hh = i >> 12, n = i & (TOTAL - 1);
  const float* a = gat_a + hh * 2 * HD;
  float ss = 0.f, dd = 0.f;
  for (int c = 0; c < HD; ++c) {
    float w = bf2f(whT[(size_t)(hh * HD + c) * TOTAL + n]);
    ss += w * a[c];
    dd += w * a[HD + c];
  }
  src[i] = ss; dst[i] = dd;
}

// ---------------- per-row softmax stats over bit-packed adjacency ----------------
// max via leaky-relu monotonicity: max_j e_ij = lrelu(src_i + max_{adj} dst_j), then sum-exp
__global__ __launch_bounds__(128) void k_stats(const unsigned int* __restrict__ abits, const float* __restrict__ src,
                                               const float* __restrict__ dst, float* __restrict__ rmax,
                                               float* __restrict__ rsinv) {
  int wave = threadIdx.x >> 5, lane = threadIdx.x & 31;
  int row = blockIdx.x * 4 + wave;
  const unsigned int* arow = abits + (size_t)row * AWORDS;
  float m0 = -3e38f, m1 = -3e38f, m2 = -3e38f, m3 = -3e38f;
  for (int j = lane; j < TOTAL; j += 32) {
    if ((arow[j >> 5] >> lane) & 1u) {
      m0 = fmaxf(m0, dst[j]);
      m1 = fmaxf(m1, dst[TOTAL + j]);
      m2 = fmaxf(m2, dst[2 * TOTAL + j]);
      m3 = fmaxf(m3, dst[3 * TOTAL + j]);
    }
  }
  for (int o = 16; o; o >>= 1) {
    m0 = fmaxf(m0, __shfl_xor(m0, o, 32)); m1 = fmaxf(m1, __shfl_xor(m1, o, 32));
    m2 = fmaxf(m2, __shfl_xor(m2, o, 32)); m3 = fmaxf(m3, __shfl_xor(m3, o, 32));
  }
  float sA0 = src[row], sA1 = src[TOTAL + row], sA2 = src[2 * TOTAL + row], sA3 = src[3 * TOTAL + row];
  float e0 = lrelu(sA0 + m0), e1 = lrelu(sA1 + m1), e2 = lrelu(sA2 + m2), e3 = lrelu(sA3 + m3);
  float s0 = 0.f, s1 = 0.f, s2 = 0.f, s3 = 0.f;
  for (int j = lane; j < TOTAL; j += 32) {
    if ((arow[j >> 5] >> lane) & 1u) {
      s0 += __expf(lrelu(sA0 + dst[j]) - e0);
      s1 += __expf(lrelu(sA1 + dst[TOTAL + j]) - e1);
      s2 += __expf(lrelu(sA2 + dst[2 * TOTAL + j]) - e2);
      s3 += __expf(lrelu(sA3 + dst[3 * TOTAL + j]) - e3);
    }
  }
  for (int o = 16; o; o >>= 1) {
    s0 += __shfl_xor(s0, o, 32); s1 += __shfl_xor(s1, o, 32);
    s2 += __shfl_xor(s2, o, 32); s3 += __shfl_xor(s3, o, 32);
  }
  if (lane == 0) {
    rmax[row] = e0; rmax[TOTAL + row] = e1; rmax[2 * TOTAL + row] = e2; rmax[3 * TOTAL + row] = e3;
    rsinv[row] = s0 > 0.f ? 1.f / s0 : 0.f;
    rsinv[TOTAL + row] = s1 > 0.f ? 1.f / s1 : 0.f;
    rsinv[2 * TOTAL + row] = s2 > 0.f ? 1.f / s2 : 0.f;
    rsinv[3 * TOTAL + row] = s3 > 0.f ? 1.f / s3 : 0.f;
  }
}

// ---------------- masked attention aggregate: h = 0.5*relu(softmax(e)*Wh) + 0.5*h ----------------
// block = 128 threads = 4 independent waves (one per head); 16 rows per block.
// Adjacency from bitmask (1 dword / 32 cols), dst broadcast via shuffle: no LDS, no barriers.
__global__ __launch_bounds__(128) void k_aggregate(const unsigned int* __restrict__ abits,
                                                   const unsigned short* __restrict__ whT,
                                                   const float* __restrict__ src, const float* __restrict__ dst,
                                                   const float* __restrict__ rmax, const float* __restrict__ rsinv,
                                                   float* __restrict__ h) {
  const int head = threadIdx.x >> 5;
  const int lane = threadIdx.x & 31;
  const int m = lane & 15, hi = lane >> 4, n = lane & 15;
  const int rowbase = blockIdx.x * 16;
  const int r0 = rowbase + m;
  const float src0 = src[head * TOTAL + r0];
  const float rm0 = rmax[head * TOTAL + r0];
  const unsigned int* arow = abits + (size_t)r0 * AWORDS;
  const float* dsth = dst + head * TOTAL;
  v8f acc[4] = {};
  for (int jt = 0; jt < TOTAL; jt += 32) {
    unsigned int am = arow[jt >> 5];            // this row's mask bits for the 32-wide tile
    float dval = dsth[jt + lane];               // lane l holds dst[jt + l]
    if (jt + 32 < TOTAL)
      __builtin_prefetch(&arow[(jt >> 5) + 1], 0, 1);
    FragBF pa;
#pragma unroll
    for (int v = 0; v < 8; ++v) {
      int k0 = 2 * v + ((v >= 4) ? 8 : 0) + 8 * hi;       // A 16x32 bf16 layout
      float d0 = __shfl(dval, k0, 32);
      float d1 = __shfl(dval, k0 + 1, 32);
      float p0 = ((am >> k0) & 1u)       ? __expf(lrelu(src0 + d0) - rm0) : 0.f;
      float p1 = ((am >> (k0 + 1)) & 1u) ? __expf(lrelu(src0 + d1) - rm0) : 0.f;
      pa.u[v] = pack2bf(p0, p1);
    }
#pragma unroll
    for (int nt = 0; nt < 4; ++nt) {
      FragBF b;
      const unsigned short* bp = whT + (size_t)(head * HD + nt * 16 + n) * TOTAL + jt + 8 * hi;
#pragma unroll
      for (int v = 0; v < 8; ++v) b.u[v] = *(const unsigned int*)(bp + 2 * v + ((v >= 4) ? 8 : 0));
      acc[nt] = __builtin_amdgcn_wmma_f32_16x16x32_bf16(false, pa.v, false, b.v, (short)0, acc[nt], false, false);
    }
  }
#pragma unroll
  for (int v = 0; v < 8; ++v) {
    int row = rowbase + v + 8 * hi;                       // C/D: M = vgpr + 8*hi
    float li = rsinv[head * TOTAL + row];
#pragma unroll
    for (int nt = 0; nt < 4; ++nt) {
      int col = head * HD + nt * 16 + n;
      size_t idx = (size_t)row * HID + col;
      h[idx] = 0.5f * fmaxf(acc[nt][v] * li, 0.f) + 0.5f * h[idx];
    }
  }
}

// ---------------- final scoring (parallel): per-sentence dot/cos, then tiny softmax ----------------
__global__ __launch_bounds__(256) void k_scores(const float* __restrict__ h, const float* __restrict__ qn,
                                                float* __restrict__ scores, float* __restrict__ cosf) {
  int wave = threadIdx.x >> 5, lane = threadIdx.x & 31;
  int j = blockIdx.x * 8 + wave;
  if (j >= NSENT) return;
  const float* qh = h;
  const float* srow = h + (size_t)(1 + MDOCS + j) * HID;
  float dot = 0.f, sn2 = 0.f;
  for (int c = lane; c < HID; c += 32) { float v = srow[c]; dot += qh[c] * v; sn2 += v * v; }
  for (int o = 16; o; o >>= 1) { dot += __shfl_xor(dot, o, 32); sn2 += __shfl_xor(sn2, o, 32); }
  if (lane == 0) {
    scores[j] = dot * 0.0625f;   // / sqrt(256)
    cosf[j] = dot / ((*qn + 1e-8f) * (sqrtf(sn2) + 1e-8f));
  }
}

__global__ __launch_bounds__(256) void k_softmax_aw(const float* __restrict__ scores, float* __restrict__ aw) {
  __shared__ float red[256];
  float sc[16];
  float lmax = -3e38f;
  for (int it = 0; it < 16; ++it) {
    int j = threadIdx.x + (it << 8);
    sc[it] = (j < NSENT) ? scores[j] : -3e38f;
    lmax = fmaxf(lmax, sc[it]);
  }
  red[threadIdx.x] = lmax; __syncthreads();
  for (int o = 128; o; o >>= 1) { if (threadIdx.x < o) red[threadIdx.x] = fmaxf(red[threadIdx.x], red[threadIdx.x + o]); __syncthreads(); }
  float gmax = red[0]; __syncthreads();
  float lsum = 0.f;
  for (int it = 0; it < 16; ++it) { int j = threadIdx.x + (it << 8); if (j < NSENT) lsum += __expf(sc[it] - gmax); }
  red[threadIdx.x] = lsum; __syncthreads();
  for (int o = 128; o; o >>= 1) { if (threadIdx.x < o) red[threadIdx.x] += red[threadIdx.x + o]; __syncthreads(); }
  float ginv = 1.f / red[0];
  for (int it = 0; it < 16; ++it) { int j = threadIdx.x + (it << 8); if (j < NSENT) aw[j] = __expf(sc[it] - gmax) * ginv; }
}

// ---------------- build fused feature matrix (bf16, K padded 514->544, M padded -> 4064) ----------------
__global__ void k_feat(const float* __restrict__ h, const float* __restrict__ aw, const float* __restrict__ cosf,
                       unsigned short* __restrict__ feat) {
  int i = blockIdx.x * 256 + threadIdx.x;
  if (i >= MPAD * FEATKP) return;
  int j = i / FEATKP, k = i % FEATKP;
  float v = 0.f;
  if (j < NSENT) {
    if (k < HID)          v = h[(size_t)(1 + MDOCS + j) * HID + k];
    else if (k < 2 * HID) v = h[k - HID] * (aw[j] + 0.5f);
    else if (k == 2 * HID)     v = cosf[j];
    else if (k == 2 * HID + 1) v = aw[j];
  }
  feat[i] = f2bf(v);
}

// ---------------- output dot ----------------
__global__ __launch_bounds__(256) void k_out(const float* __restrict__ g, const float* __restrict__ out_w,
                                             const float* __restrict__ out_b, const float* __restrict__ cosf,
                                             float* __restrict__ out) {
  int wave = threadIdx.x >> 5, lane = threadIdx.x & 31;
  int j = blockIdx.x * 8 + wave;
  if (j >= NSENT) return;
  const float* grow = g + (size_t)j * HID;
  float s = 0.f;
  for (int c = lane; c < HID; c += 32) s += grow[c] * out_w[c];
  for (int o = 16; o; o >>= 1) s += __shfl_xor(s, o, 32);
  if (lane == 0) out[j] = s + out_b[0] + 0.5f * cosf[j];
}

extern "C" void kernel_launch(void* const* d_in, const int* in_sizes, int n_in,
                              void* d_out, int out_size, void* d_ws, size_t ws_size,
                              hipStream_t stream) {
  (void)in_sizes; (void)n_in; (void)out_size; (void)ws_size;
  const float* emb      = (const float*)d_in[0];
  const int*   adj      = (const int*)d_in[1];
  const float* proj_w   = (const float*)d_in[3];
  const float* proj_b   = (const float*)d_in[4];
  const float* ln_scale = (const float*)d_in[5];
  const float* ln_bias  = (const float*)d_in[6];
  const float* gat_W    = (const float*)d_in[7];
  const float* gat_a    = (const float*)d_in[8];
  const float* fusion_w = (const float*)d_in[9];
  const float* fusion_b = (const float*)d_in[10];
  const float* out_w    = (const float*)d_in[11];
  const float* out_b    = (const float*)d_in[12];
  float* out = (float*)d_out;

  char* ws = (char*)d_ws;
  size_t off = 0;
  auto take = [&](size_t bytes) -> char* {
    char* p = ws + off;
    off += (bytes + 255) & ~(size_t)255;
    return p;
  };
  unsigned short* emb_bf   = (unsigned short*)take((size_t)TOTAL * EMB * 2);
  unsigned short* projw_bf = (unsigned short*)take((size_t)HID * EMB * 2);
  unsigned short* gatw_bf  = (unsigned short*)take((size_t)2 * HID * HID * 2);
  unsigned short* fusw_bf  = (unsigned short*)take((size_t)HID * FEATKP * 2);
  unsigned int*   abits    = (unsigned int*)take((size_t)TOTAL * AWORDS * 4);
  float*          h        = (float*)take((size_t)TOTAL * HID * 4);
  unsigned short* x_bf     = (unsigned short*)take((size_t)TOTAL * HID * 2);
  unsigned short* whT      = (unsigned short*)take((size_t)HID * TOTAL * 2);
  float*          srcv     = (float*)take((size_t)HEADS * TOTAL * 4);
  float*          dstv     = (float*)take((size_t)HEADS * TOTAL * 4);
  float*          rmaxv    = (float*)take((size_t)HEADS * TOTAL * 4);
  float*          rsinv    = (float*)take((size_t)HEADS * TOTAL * 4);
  float*          qn       = (float*)take(256);
  float*          scores   = (float*)take((size_t)MPAD * 4);
  float*          aw       = (float*)take((size_t)MPAD * 4);
  float*          cosf     = (float*)take((size_t)MPAD * 4);
  unsigned short* feat     = (unsigned short*)take((size_t)MPAD * FEATKP * 2);
  float*          g        = (float*)take((size_t)MPAD * HID * 4);

  // conversions + adjacency bit-pack (adj touched once; later passes read the 2MB bitmask)
  k_f32_to_bf16<<<(TOTAL * EMB + 255) / 256, 256, 0, stream>>>(emb, emb_bf, TOTAL * EMB);
  k_f32_to_bf16<<<(HID * EMB + 255) / 256, 256, 0, stream>>>(proj_w, projw_bf, HID * EMB);
  k_f32_to_bf16<<<(2 * HID * HID + 255) / 256, 256, 0, stream>>>(gat_W, gatw_bf, 2 * HID * HID);
  k_pad_fusionw<<<(HID * FEATKP + 255) / 256, 256, 0, stream>>>(fusion_w, fusw_bf);
  k_packadj<<<(TOTAL * TOTAL) / 256, 256, 0, stream>>>(adj, abits);

  // h = relu(emb @ proj_w^T + b)
  wmma_gemm_bt<<<dim3(TOTAL / 16, HID / 64), 128, 0, stream>>>(
      emb_bf, projw_bf, proj_b, h, (unsigned short*)nullptr, TOTAL, HID, EMB, 1);

  // query-cos blend of sentence rows
  k_qnorm<<<1, 256, 0, stream>>>(h, qn);
  k_blend<<<(NSENT + 7) / 8, 256, 0, stream>>>(h, qn);

  for (int l = 0; l < 2; ++l) {
    k_layernorm<<<TOTAL / 8, 256, 0, stream>>>(h, ln_scale + l * HID, ln_bias + l * HID, x_bf);
    // WhT (256 x 4096, bf16 transposed) = (x @ Wcat^T)^T
    wmma_gemm_bt<<<dim3(TOTAL / 16, HID / 64), 128, 0, stream>>>(
        x_bf, gatw_bf + (size_t)l * HID * HID, nullptr, (float*)nullptr, whT, TOTAL, HID, HID, 0);
    k_srcdst<<<(HEADS * TOTAL) / 256, 256, 0, stream>>>(whT, gat_a + l * HEADS * 2 * HD, srcv, dstv);
    k_stats<<<TOTAL / 4, 128, 0, stream>>>(abits, srcv, dstv, rmaxv, rsinv);
    k_aggregate<<<TOTAL / 16, 128, 0, stream>>>(abits, whT, srcv, dstv, rmaxv, rsinv, h);
  }

  // final scoring softmax + cos (parallel dot pass, tiny single-block softmax)
  k_qnorm<<<1, 256, 0, stream>>>(h, qn);
  k_scores<<<(NSENT + 7) / 8, 256, 0, stream>>>(h, qn, scores, cosf);
  k_softmax_aw<<<1, 256, 0, stream>>>(scores, aw);

  k_feat<<<(MPAD * FEATKP + 255) / 256, 256, 0, stream>>>(h, aw, cosf, feat);
  wmma_gemm_bt<<<dim3(MPAD / 16, HID / 64), 128, 0, stream>>>(
      feat, fusw_bf, fusion_b, g, (unsigned short*)nullptr, MPAD, HID, FEATKP, 1);
  k_out<<<(NSENT + 7) / 8, 256, 0, stream>>>(g, out_w, out_b, cosf, out);
}
